// AttentionBlock_89232240542315
// MI455X (gfx1250) — matrix operand order repaired and checked by reading, not compiled, not run on hardware
//
#include <hip/hip_runtime.h>

// Problem constants (match reference)
#define BB   32
#define CC   512
#define LL   4096
#define HH   8
#define DD   64
#define GG   32
#define CPG  (CC / GG)          // 16 channels per group
#define MM   ((size_t)BB * LL)  // 131072 rows
#define EPSV 1e-5f
#define NEGV -1e30f

typedef __attribute__((ext_vector_type(16))) _Float16 v16h;
typedef __attribute__((ext_vector_type(8)))  _Float16 v8h;
typedef __attribute__((ext_vector_type(8)))  float    v8f;

// LDS row padding for the staged A tile: 512 + 8 halves -> 1040B row stride,
// 16B aligned and bank-conflict-free across rows.
#define APAD 520

// ---------------------------------------------------------------------------
// 1) GroupNorm statistics: one block per (b,g); group slice is a contiguous
//    CPG*L = 65536-float segment of x (B,C,L layout).
// ---------------------------------------------------------------------------
__global__ __launch_bounds__(256) void gn_stats_kernel(
    const float* __restrict__ x, float* __restrict__ stats) {
  const int bg = blockIdx.x;                       // b*G + g
  const float* p = x + (size_t)bg * (CPG * LL);
  float s = 0.f, ss = 0.f;
  for (int i = threadIdx.x; i < CPG * LL; i += 256) {
    float v = p[i];
    s += v; ss += v * v;
  }
  __shared__ float sh_s[256];
  __shared__ float sh_q[256];
  sh_s[threadIdx.x] = s; sh_q[threadIdx.x] = ss;
  __syncthreads();
  for (int st = 128; st > 0; st >>= 1) {
    if (threadIdx.x < st) {
      sh_s[threadIdx.x] += sh_s[threadIdx.x + st];
      sh_q[threadIdx.x] += sh_q[threadIdx.x + st];
    }
    __syncthreads();
  }
  if (threadIdx.x == 0) {
    const float n = (float)(CPG * LL);
    float mu = sh_s[0] / n;
    float var = sh_q[0] / n - mu * mu;
    stats[2 * bg + 0] = mu;
    stats[2 * bg + 1] = rsqrtf(var + EPSV);
  }
}

// ---------------------------------------------------------------------------
// 2) Normalize + transpose: (B,C,L) f32 -> (B*L, C) f16, LDS-tiled 32x32.
//    grid (L/32, C/32, B), block (32,8)
// ---------------------------------------------------------------------------
__global__ __launch_bounds__(256) void gn_apply_transpose_kernel(
    const float* __restrict__ x, const float* __restrict__ gamma,
    const float* __restrict__ beta, const float* __restrict__ stats,
    _Float16* __restrict__ hbuf) {
  __shared__ float tile[32][33];
  const int b  = blockIdx.z;
  const int l0 = blockIdx.x * 32;
  const int c0 = blockIdx.y * 32;
#pragma unroll
  for (int i = 0; i < 4; ++i) {
    const int cl = threadIdx.y + i * 8;
    const int c  = c0 + cl;
    const int l  = l0 + threadIdx.x;
    const int g  = c / CPG;
    const float mu = stats[2 * (b * GG + g) + 0];
    const float rs = stats[2 * (b * GG + g) + 1];
    const float v  = x[((size_t)b * CC + c) * LL + l];
    tile[cl][threadIdx.x] = (v - mu) * rs * gamma[c] + beta[c];
  }
  __syncthreads();
#pragma unroll
  for (int i = 0; i < 4; ++i) {
    const int ll = threadIdx.y + i * 8;
    const int l  = l0 + ll;
    const int c  = c0 + threadIdx.x;
    hbuf[((size_t)b * LL + l) * CC + c] = (_Float16)tile[threadIdx.x][ll];
  }
}

// ---------------------------------------------------------------------------
// 3) Weight convert: W (K x N row-major f32) -> Wt (N-major f16, Wt[n*512+k]).
//    Three matrices packed back-to-back: Wk, Wv, Wo.
// ---------------------------------------------------------------------------
__global__ __launch_bounds__(256) void weight_convert_kernel(
    const float* __restrict__ Wk, const float* __restrict__ Wv,
    const float* __restrict__ Wo, _Float16* __restrict__ Wt) {
  const int idx = blockIdx.x * 256 + threadIdx.x;  // 3 * 512 * 512 total
  const int mat = idx >> 18;
  const int r   = idx & 262143;
  const int n   = r >> 9;
  const int k   = r & 511;
  const float* W = (mat == 0) ? Wk : ((mat == 1) ? Wv : Wo);
  Wt[idx] = (_Float16)W[k * 512 + n];
}

// ---------------------------------------------------------------------------
// WMMA fragment helpers (contiguous k order k = 16*half + s for both A and B, wave32)
// ---------------------------------------------------------------------------
__device__ __forceinline__ v16h make_frag(v8h lo, v8h hi) {
  return __builtin_shufflevector(lo, hi, 0, 1, 2, 3, 4, 5, 6, 7,
                                 8, 9, 10, 11, 12, 13, 14, 15);
}

// A fragment from the LDS-staged tile.
// lane<16: K = kb+0..15 ; lane>=16: K = kb+16..31
__device__ __forceinline__ v16h load_a_frag_lds(
    const _Float16* __restrict__ Arow, int kb, int half) {
  const int ka = kb + (half ? 16 : 0);
  v8h lo = *(const v8h*)(Arow + ka);
  v8h hi = *(const v8h*)(Arow + ka + 8);
  return make_frag(lo, hi);
}

// B fragment from N-major weights. Bcol = &Wt[n*512].
// lane<16: K = kb+0..15 ; lane>=16: K = kb+16..31
__device__ __forceinline__ v16h load_b_frag(
    const _Float16* __restrict__ Bcol, int kb, int half) {
  const int kc = kb + (half ? 16 : 0);
  v8h lo = *(const v8h*)(Bcol + kc);
  v8h hi = *(const v8h*)(Bcol + kc + 8);
  return make_frag(lo, hi);
}

// Stage a ROWS x 512 f16 A tile into LDS via CDNA5 async load-to-LDS.
// ROWS*64 16-byte chunks, 256 threads.
template <int ROWS>
__device__ __forceinline__ void stage_a_tile_async(
    _Float16 (*As)[APAD], const _Float16* __restrict__ Aglob /* row0 of tile */) {
  const int tid = threadIdx.x;
#pragma unroll
  for (int it = 0; it < ROWS * 64 / 256; ++it) {
    const int c   = tid + it * 256;       // chunk id
    const int row = c >> 6;               // 64 chunks of 8 halves per row
    const int col = (c & 63) * 8;
    const unsigned lds_addr = (unsigned)(uintptr_t)&As[row][col];
    const _Float16* gaddr = Aglob + (size_t)row * CC + col;
    asm volatile("global_load_async_to_lds_b128 %0, %1, off"
                 :: "v"(lds_addr), "v"(gaddr)
                 : "memory");
  }
  asm volatile("s_wait_asynccnt 0x0" ::: "memory");
  __syncthreads();
}

// ---------------------------------------------------------------------------
// 4) KV GEMM: kv[m, n] = hbuf[m, :] @ Wt[n, :], N = 1024 (k then v).
//    Block = 8 waves; block owns TWO 16-row m-tiles (32 rows) staged in LDS;
//    wave w covers columns [w*128, w*128+128). Each B fragment feeds two
//    WMMAs (one per m-tile) -> wmma:vmem ~1. grid (M/32), block 256.
// ---------------------------------------------------------------------------
__global__ __launch_bounds__(256) void gemm_kv_kernel(
    const _Float16* __restrict__ hbuf, const _Float16* __restrict__ Wt,
    _Float16* __restrict__ kv) {
  __shared__ _Float16 As[32][APAD];
  const int m0 = blockIdx.x * 32;
  stage_a_tile_async<32>(As, hbuf + (size_t)m0 * CC);

  const int wave   = threadIdx.x >> 5;
  const int lane   = threadIdx.x & 31;
  const int half   = lane >> 4;
  const int lane16 = lane & 15;
  const int n0w    = wave * 128;

  v8f accL[8] = {}, accH[8] = {};
  const _Float16* ArowL = &As[lane16][0];
  const _Float16* ArowH = &As[16 + lane16][0];
  const _Float16* B0    = Wt + (size_t)(n0w + lane16) * CC;

#pragma unroll
  for (int kb = 0; kb < CC; kb += 32) {
    const v16h a0 = load_a_frag_lds(ArowL, kb, half);
    const v16h a1 = load_a_frag_lds(ArowH, kb, half);
#pragma unroll
    for (int j = 0; j < 8; ++j) {
      const v16h b = load_b_frag(B0 + j * 16 * CC, kb, half);
      accL[j] = __builtin_amdgcn_wmma_f32_16x16x32_f16(false, a0, false, b, (short)0, accL[j], false, false);
      accH[j] = __builtin_amdgcn_wmma_f32_16x16x32_f16(false, a1, false, b, (short)0, accH[j], false, false);
    }
  }
#pragma unroll
  for (int j = 0; j < 8; ++j) {
    const int n = n0w + j * 16 + lane16;
#pragma unroll
    for (int i = 0; i < 8; ++i) {
      const int mr = i + 8 * half;
      kv[(size_t)(m0 + mr) * 1024 + n]      = (_Float16)accL[j][i];
      kv[(size_t)(m0 + 16 + mr) * 1024 + n] = (_Float16)accH[j][i];
    }
  }
}

// ---------------------------------------------------------------------------
// 5) Logits: logits[m, h] = (k[m, h, :] . q[h, :]) / sqrt(D)
// ---------------------------------------------------------------------------
__global__ __launch_bounds__(256) void logits_kernel(
    const _Float16* __restrict__ kv, const float* __restrict__ q,
    float* __restrict__ logits) {
  const int idx = blockIdx.x * 256 + threadIdx.x;  // m*H + h
  const int m  = idx >> 3;
  const int h  = idx & 7;
  const _Float16* kp = kv + (size_t)m * 1024 + h * DD;
  const float*    qp = q + h * DD;
  float s = 0.f;
#pragma unroll 8
  for (int d = 0; d < DD; ++d) s += (float)kp[d] * qp[d];
  logits[idx] = s * 0.125f;  // 1/sqrt(64)
}

// ---------------------------------------------------------------------------
// 6) Window-3 softmax + weighted V:  o[m, h*64+d]
// ---------------------------------------------------------------------------
__global__ __launch_bounds__(256) void attn_kernel(
    const _Float16* __restrict__ kv, const float* __restrict__ logits,
    const float* __restrict__ rpe, _Float16* __restrict__ obuf) {
  const int idx = blockIdx.x * 256 + threadIdx.x;  // m*C + n
  const int m = idx >> 9;
  const int n = idx & 511;
  const int h = n >> 6;
  const int l = m & (LL - 1);
  const int base = m - l;  // b*L

  float w0 = (l - 1 >= 0) ? logits[(size_t)(base + l - 1) * HH + h] + rpe[h * 3 + 0] : NEGV;
  float w1 =                logits[(size_t)m * HH + h]              + rpe[h * 3 + 1];
  float w2 = (l + 1 < LL) ? logits[(size_t)(base + l + 1) * HH + h] + rpe[h * 3 + 2] : NEGV;

  const float mx = fmaxf(w0, fmaxf(w1, w2));
  const float e0 = __expf(w0 - mx);
  const float e1 = __expf(w1 - mx);
  const float e2 = __expf(w2 - mx);
  const float inv = 1.f / (e0 + e1 + e2);

  float acc = e1 * inv * (float)kv[(size_t)m * 1024 + 512 + n];
  if (l - 1 >= 0) acc += e0 * inv * (float)kv[(size_t)(base + l - 1) * 1024 + 512 + n];
  if (l + 1 < LL) acc += e2 * inv * (float)kv[(size_t)(base + l + 1) * 1024 + 512 + n];
  obuf[idx] = (_Float16)acc;
}

// ---------------------------------------------------------------------------
// 7) Output GEMM + residual, scatter back to (B, C, L):
//    out[b, n, l] = x[b, n, l] + (obuf @ Wo)[m=b*L+l, n]
//    Block = 8 waves; TWO 16-row m-tiles staged in LDS; wave w covers
//    columns [w*64, w*64+64). grid (M/32), block 256.
// ---------------------------------------------------------------------------
__global__ __launch_bounds__(256) void gemm_o_kernel(
    const _Float16* __restrict__ obuf, const _Float16* __restrict__ Wot,
    const float* __restrict__ x, float* __restrict__ out) {
  __shared__ _Float16 As[32][APAD];
  const int m0 = blockIdx.x * 32;
  stage_a_tile_async<32>(As, obuf + (size_t)m0 * CC);

  const int wave   = threadIdx.x >> 5;
  const int lane   = threadIdx.x & 31;
  const int half   = lane >> 4;
  const int lane16 = lane & 15;
  const int n0w    = wave * 64;

  v8f accL[4] = {}, accH[4] = {};
  const _Float16* ArowL = &As[lane16][0];
  const _Float16* ArowH = &As[16 + lane16][0];
  const _Float16* B0    = Wot + (size_t)(n0w + lane16) * CC;

#pragma unroll
  for (int kb = 0; kb < CC; kb += 32) {
    const v16h a0 = load_a_frag_lds(ArowL, kb, half);
    const v16h a1 = load_a_frag_lds(ArowH, kb, half);
#pragma unroll
    for (int j = 0; j < 4; ++j) {
      const v16h b = load_b_frag(B0 + j * 16 * CC, kb, half);
      accL[j] = __builtin_amdgcn_wmma_f32_16x16x32_f16(false, a0, false, b, (short)0, accL[j], false, false);
      accH[j] = __builtin_amdgcn_wmma_f32_16x16x32_f16(false, a1, false, b, (short)0, accH[j], false, false);
    }
  }
#pragma unroll
  for (int j = 0; j < 4; ++j) {
    const int n = n0w + j * 16 + lane16;
#pragma unroll
    for (int i = 0; i < 8; ++i) {
      const int mr = i + 8 * half;
      {
        const int m = m0 + mr;
        const size_t p = ((size_t)(m >> 12) * CC + n) * LL + (m & (LL - 1));
        out[p] = x[p] + accL[j][i];
      }
      {
        const int m = m0 + 16 + mr;
        const size_t p = ((size_t)(m >> 12) * CC + n) * LL + (m & (LL - 1));
        out[p] = x[p] + accH[j][i];
      }
    }
  }
}

// ---------------------------------------------------------------------------
// Launcher
// ---------------------------------------------------------------------------
extern "C" void kernel_launch(void* const* d_in, const int* in_sizes, int n_in,
                              void* d_out, int out_size, void* d_ws, size_t ws_size,
                              hipStream_t stream) {
  const float* x     = (const float*)d_in[0];
  const float* gamma = (const float*)d_in[1];
  const float* beta  = (const float*)d_in[2];
  const float* W_k   = (const float*)d_in[3];
  const float* W_v   = (const float*)d_in[4];
  const float* W_o   = (const float*)d_in[5];
  const float* q     = (const float*)d_in[6];
  const float* rpe   = (const float*)d_in[7];
  float* out = (float*)d_out;

  // Workspace carve-up (aligned)
  char* wsb = (char*)d_ws;
  float*     stats = (float*)wsb;                                   // 2*B*G f32
  size_t off = 16384;
  _Float16*  Wt    = (_Float16*)(wsb + off); off += 3ull * 512 * 512 * 2;
  off = (off + 255) & ~(size_t)255;
  _Float16*  hbuf  = (_Float16*)(wsb + off); off += MM * CC * 2;    // 128 MiB
  _Float16*  kvbuf = (_Float16*)(wsb + off); off += MM * 1024 * 2;  // 256 MiB
  float*     lgt   = (float*)(wsb + off);    off += MM * HH * 4;    //   4 MiB
  _Float16*  obuf  = (_Float16*)(wsb + off); off += MM * CC * 2;    // 128 MiB
  (void)ws_size; (void)in_sizes; (void)n_in; (void)out_size;

  // 1) GroupNorm stats
  gn_stats_kernel<<<dim3(BB * GG), dim3(256), 0, stream>>>(x, stats);

  // 2) Normalize + transpose -> h f16 (B*L, C)
  gn_apply_transpose_kernel<<<dim3(LL / 32, CC / 32, BB), dim3(32, 8), 0, stream>>>(
      x, gamma, beta, stats, hbuf);

  // 3) Weights -> N-major f16
  weight_convert_kernel<<<dim3(3 * 512 * 512 / 256), dim3(256), 0, stream>>>(
      W_k, W_v, W_o, Wt);

  // 4) K/V projection GEMM (N = 1024 fused), 32-row A tile staged async->LDS
  gemm_kv_kernel<<<dim3((unsigned)(MM / 32)), dim3(256), 0, stream>>>(
      hbuf, Wt, kvbuf);

  // 5) Logits
  logits_kernel<<<dim3((unsigned)(MM * HH / 256)), dim3(256), 0, stream>>>(
      kvbuf, q, lgt);

  // 6) Window softmax + weighted V
  attn_kernel<<<dim3((unsigned)(MM * CC / 256)), dim3(256), 0, stream>>>(
      kvbuf, lgt, rpe, obuf);

  // 7) Output projection + residual (scatter to (B,C,L))
  gemm_o_kernel<<<dim3((unsigned)(MM / 32)), dim3(256), 0, stream>>>(
      obuf, Wt + 2ull * 512 * 512, x, out);
}
